// ShiftedL2Loss_8495445311687
// MI455X (gfx1250) — matrix operand order repaired and checked by reading, not compiled, not run on hardware
//
#include <hip/hip_runtime.h>
#include <math.h>

// ---- problem constants (from reference) ----
constexpr int ROWS = 32;          // 4*8
constexpr int L    = 15104;
constexpr int T    = 128;         // T_WIN
constexpr int NWIN = 118;         // (L - T)/DT + 1
constexpr int LI   = L - 2 * T;   // 14848 interior elements
constexpr int KPAD = 384;         // K padded (need >= 368)
constexpr int XOFF = 240;         // lds_xh offset: j in [-240, 384)
constexpr int XPAD = 624;         // 240 + 384 (halfs)
constexpr float INV_T   = 1.0f / 128.0f;
constexpr float INV_CNT = 1.0f / (float)(ROWS * NWIN); // 1/3776

typedef __attribute__((ext_vector_type(16))) _Float16 v16h;
typedef __attribute__((ext_vector_type(8)))  _Float16 v8h;
typedef __attribute__((ext_vector_type(4)))  _Float16 v4h;
typedef __attribute__((ext_vector_type(8)))  float    v8f;

__device__ __forceinline__ float wave_sum(float v) {
#pragma unroll
    for (int o = 16; o > 0; o >>= 1) v += __shfl_xor(v, o, 32);
    return v;
}
__device__ __forceinline__ float wave_min(float v) {
#pragma unroll
    for (int o = 16; o > 0; o >>= 1) v = fminf(v, __shfl_xor(v, o, 32));
    return v;
}

__global__ void zero_out_kernel(float* out) {
    if (threadIdx.x == 0 && blockIdx.x == 0) out[0] = 0.0f;
}

// One block per row: inv_norm[row] = 1 / (sqrt(min(mag_x,mag_r)*mag_r) + 1e-3)
__global__ void row_stats_kernel(const float* __restrict__ x,
                                 const float* __restrict__ r,
                                 float* __restrict__ inv_norm) {
    const int row = blockIdx.x;
    const float* xr = x + (size_t)row * L;
    const float* rr = r + (size_t)row * L;
    float sd = 0.f, sd2 = 0.f, sr = 0.f, sr2 = 0.f;
    for (int i = T + (int)threadIdx.x; i < L - T; i += (int)blockDim.x) {
        float a = xr[i], b = rr[i];
        sd += a; sd2 += a * a; sr += b; sr2 += b * b;
    }
    sd = wave_sum(sd); sd2 = wave_sum(sd2); sr = wave_sum(sr); sr2 = wave_sum(sr2);
    __shared__ float red[4][8];
    const int wid = threadIdx.x >> 5, ln = threadIdx.x & 31;
    if (ln == 0) { red[0][wid] = sd; red[1][wid] = sd2; red[2][wid] = sr; red[3][wid] = sr2; }
    __syncthreads();
    if (threadIdx.x == 0) {
        float a = 0.f, b = 0.f, c = 0.f, d = 0.f;
#pragma unroll
        for (int i = 0; i < 8; ++i) { a += red[0][i]; b += red[1][i]; c += red[2][i]; d += red[3][i]; }
        const float invLi = 1.0f / (float)LI;
        float md = a * invLi, mr = c * invLi;
        float vx = fmaxf(b * invLi - md * md, 0.f);
        float vr = fmaxf(d * invLi - mr * mr, 0.f);
        float magx = sqrtf(vx), magr = sqrtf(vr);
        float magnorm = sqrtf(fminf(magx, magr) * magr) + 0.001f;
        inv_norm[row] = 1.0f / magnorm;
    }
}

// One 32-lane wave per (row, window). Shift search via WMMA, s = 16m + n:
//   dot[16m+n] = sum_j A[m,j]*B[j,n],  A[m,j]=x[j-16m] (zero-padded f16), B[j,n]=rp[base+n+j]
// Sliding energy via f32 prefix sums: sy2[s] = P[s+128] - P[s].
__global__ void shifted_l2_kernel(const float* __restrict__ x,
                                  const float* __restrict__ r,
                                  const float* __restrict__ inv_norm,
                                  float* __restrict__ out) {
    const int row  = blockIdx.x / NWIN;
    const int win  = blockIdx.x % NWIN;
    const int lane = threadIdx.x;

    __shared__ _Float16 lds_xh[XPAD]; // x window as f16, zero-padded: valid [XOFF, XOFF+128)
    __shared__ float    lds_y[400];   // normalized rp[base + jj], jj in [0,400)
    __shared__ float    lds_P[385];   // prefix sums of y^2 over [0,384]

    const float inv = inv_norm[row];
    const float* xr = x + (size_t)row * L;
    const float* rr = r + (size_t)row * L;

    __builtin_prefetch(xr + win * T, 0, 1);  // global_prefetch_b8

    // zero-fill padded x array (32-bit chunks)
    {
        unsigned* xz = (unsigned*)lds_xh;
#pragma unroll
        for (int jj = lane; jj < XPAD / 2; jj += 32) xz[jj] = 0u;
    }
    // stage x window: one float4 per lane (128 = 32*4), convert once, ||x||^2 on the fly
    float sx2;
    {
        const float4 v = *(const float4*)(xr + win * T + 4 * lane);
        const float f0 = v.x * inv, f1 = v.y * inv, f2 = v.z * inv, f3 = v.w * inv;
        v4h hv = {(_Float16)f0, (_Float16)f1, (_Float16)f2, (_Float16)f3};
        *(v4h*)(&lds_xh[XOFF + 4 * lane]) = hv;
        sx2 = wave_sum(f0 * f0 + f1 * f1 + f2 * f2 + f3 * f3);
    }

    // stage rp segment: rp[base+jj] = r[(win-1)*128 + jj], zero outside [0,L)
    const int rbase = (win - 1) * T;
    for (int jj = lane; jj < 400; jj += 32) {
        const int gi = rbase + jj;
        lds_y[jj] = (gi >= 0 && gi < L) ? rr[gi] * inv : 0.0f;
    }
    __syncthreads();  // single-wave WG: barrier -> S_NOP + LDS wait

    // ---- prefix sums of y^2: each lane owns a 12-element chunk (384 = 32*12) ----
    {
        float ysq[12];
        float csum = 0.f;
#pragma unroll
        for (int i = 0; i < 12; ++i) {
            const float v = lds_y[12 * lane + i];
            ysq[i] = v * v;
            csum += ysq[i];
        }
        // inclusive wave scan of chunk sums -> exclusive base for this lane
        float incl = csum;
#pragma unroll
        for (int o = 1; o < 32; o <<= 1) {
            const float t = __shfl_up(incl, o, 32);
            if (lane >= o) incl += t;
        }
        float run = incl - csum;  // exclusive prefix
#pragma unroll
        for (int i = 0; i < 12; ++i) {
            lds_P[12 * lane + i] = run;
            run += ysq[i];
        }
        if (lane == 31) lds_P[384] = run;  // total
    }
    __syncthreads();

    // ---- WMMA correlation: 12 x v_wmma_f32_16x16x32_f16, B gathers pipelined 1 step ----
    const int half16 = lane >> 4;   // 0 | 1
    const int mrow   = lane & 15;   // A row (coarse shift m)
    const int ncol   = lane & 15;   // B col (fine shift n)
    const int abase  = XOFF + (half16 << 4) - 16 * mrow;  // half-index, mult. of 8
    const int bbase  = (half16 << 4) + ncol;              // B gather base in lds_y

    float yb[16];  // B floats for the *current* step (prefetched one step ahead)
#pragma unroll
    for (int e = 0; e < 16; ++e) yb[e] = lds_y[bbase + e];

    v8f acc = {0.f, 0.f, 0.f, 0.f, 0.f, 0.f, 0.f, 0.f};
#pragma unroll
    for (int K0 = 0; K0 < KPAD; K0 += 32) {
        // A: two aligned 16B LDS loads, no conversion
        const v8h lo = *(const v8h*)(&lds_xh[abase + K0]);
        const v8h hi = *(const v8h*)(&lds_xh[abase + K0 + 8]);
        const v16h a = __builtin_shufflevector(lo, hi, 0, 1, 2, 3, 4, 5, 6, 7,
                                               8, 9, 10, 11, 12, 13, 14, 15);
        // convert this step's B from registers (loaded last iteration)
        v16h b;
#pragma unroll
        for (int e = 0; e < 16; ++e) b[e] = (_Float16)yb[e];

        // prefetch next step's B floats; latency hides behind the WMMA below
        if (K0 + 32 < KPAD) {
#pragma unroll
            for (int e = 0; e < 16; ++e) yb[e] = lds_y[bbase + K0 + 32 + e];
        }

        acc = __builtin_amdgcn_wmma_f32_16x16x32_f16(
            false, a, false, b, (short)0, acc, false, false);
    }

    // fuse energy + weight + min over shifts s = 16m + n (C layout: VGPR i -> M=i+8*half16)
    constexpr float TWO_PI_256  = 6.28318530717958647692f / 256.0f;
    constexpr float FOUR_PI_256 = 12.5663706143591729539f / 256.0f;
    float lmin = 3.4e38f;
#pragma unroll
    for (int i = 0; i < 8; ++i) {
        const int m = i + (half16 << 3);
        const int s = 16 * m + ncol;
        const float sy2 = lds_P[s + 128] - lds_P[s];
        const float mse = (sx2 + sy2 - 2.0f * acc[i]) * INV_T;
        const float bw  = 0.42f - 0.5f * __cosf(TWO_PI_256 * (float)s)
                                + 0.08f * __cosf(FOUR_PI_256 * (float)s);
        const float ws  = 100.0f * (1.0f - bw);
        const float lo  = (mse + 1.0f) * (ws + 1.0f) - 1.0f;
        lmin = fminf(lmin, lo);
    }

    // shift s = 256 (w = 100): wave-reduced dot, energy from prefix sums
    float d256 = 0.f;
#pragma unroll
    for (int k = lane; k < 128; k += 32) {
        d256 += (float)lds_xh[XOFF + k] * lds_y[256 + k];
    }
    d256 = wave_sum(d256);
    {
        const float sy2 = lds_P[384] - lds_P[256];
        const float mse = (sx2 + sy2 - 2.0f * d256) * INV_T;
        const float lo  = (mse + 1.0f) * 101.0f - 1.0f;
        lmin = fminf(lmin, lo);
    }

    lmin = wave_min(lmin);
    if (lane == 0) atomicAdd(out, lmin * INV_CNT);
}

extern "C" void kernel_launch(void* const* d_in, const int* in_sizes, int n_in,
                              void* d_out, int out_size, void* d_ws, size_t ws_size,
                              hipStream_t stream) {
    (void)in_sizes; (void)n_in; (void)out_size; (void)ws_size;
    const float* data   = (const float*)d_in[0];
    const float* target = (const float*)d_in[1];
    float* out      = (float*)d_out;
    float* inv_norm = (float*)d_ws;  // 32 floats of scratch

    zero_out_kernel<<<1, 32, 0, stream>>>(out);
    row_stats_kernel<<<ROWS, 256, 0, stream>>>(data, target, inv_norm);
    shifted_l2_kernel<<<ROWS * NWIN, 32, 0, stream>>>(data, target, inv_norm, out);
}
